// PatternFFN_40596030882577
// MI455X (gfx1250) — matrix operand order repaired and checked by reading, not compiled, not run on hardware
//
#include <hip/hip_runtime.h>
#include <math.h>

// ---------------------------------------------------------------------------
// PatternFFN on MI455X (gfx1250, wave32, WMMA bf16 16x16x32)
// LDS-staged GEMMs with async Global->LDS DMA (ASYNCcnt) double buffering.
// ---------------------------------------------------------------------------

typedef __attribute__((ext_vector_type(16))) __bf16 v16bf;
typedef __attribute__((ext_vector_type(8)))  float  v8f;

#define LDS_PITCH 40   // shorts per 32-K row (32 data + 8 pad) = 80B:
                       // 16B-aligned b128 reads, bank stride 20 -> conflict-free
#define AB_STRIDE  (128 * LDS_PITCH)      // shorts: A region -> B region
#define BUF_STRIDE (2 * AB_STRIDE)        // shorts: buffer 0 -> buffer 1

union Frag32 {            // 32 bytes = 16 bf16 = one WMMA A/B operand per lane
    uint4 u[2];
    v16bf v;
};

static __device__ __forceinline__ unsigned short f32_to_bf16(float f) {
    unsigned int u = __float_as_uint(f);
    u += 0x7FFFu + ((u >> 16) & 1u);      // round-to-nearest-even
    return (unsigned short)(u >> 16);
}

static __device__ __forceinline__ v8f wmma_bf16(const v16bf a, const v16bf b, const v8f c) {
    // (neg_a, A, neg_b, B, c_mod, C, reuse_a, reuse_b)
    return __builtin_amdgcn_wmma_f32_16x16x32_bf16(false, a, false, b, (short)0, c,
                                                   false, false);
}

// --- CDNA5 async Global->LDS copy (16B per lane), tracked by ASYNCcnt -------
static __device__ __forceinline__ void async_copy16(const unsigned short* g,
                                                    unsigned short* l) {
    unsigned int loff = (unsigned int)(size_t)l;   // flat -> LDS offset = addr[31:0]
    asm volatile("global_load_async_to_lds_b128 %0, %1, off"
                 :: "v"(loff), "v"(g) : "memory");
}
static __device__ __forceinline__ void wait_async0() {
    asm volatile("s_wait_asynccnt 0x0" ::: "memory");
}

// Stage one 128x32 A tile + 128x32 B tile into LDS (4 async b128 per thread).
static __device__ __forceinline__ void stage_tile(
        unsigned short* lbuf,                    // buffer base (A at 0, B at AB_STRIDE)
        const unsigned short* __restrict__ gA,   // blockM row 0, [*, K]
        const unsigned short* __restrict__ gB,   // blockN row 0, [*, K]
        int tid, int K, int k) {
    const int row  = tid >> 1;          // 0..127
    const int half = tid & 1;           // 16-short half of the 32-K row
    const unsigned short* ga = gA + (size_t)row * K + k + half * 16;
    const unsigned short* gb = gB + (size_t)row * K + k + half * 16;
    unsigned short* la = lbuf + row * LDS_PITCH + half * 16;
    unsigned short* lb = la + AB_STRIDE;
    async_copy16(ga,     la);
    async_copy16(ga + 8, la + 8);
    async_copy16(gb,     lb);
    async_copy16(gb + 8, lb + 8);
}

// 12 conflict-free ds_load_b128 + 8 WMMAs on a staged 32-K step.
static __device__ __forceinline__ void compute_step(
        v8f acc[2][4],
        const unsigned short* lbuf,
        int waveM, int waveN, int l16, int lhalf) {
    Frag32 a0, a1, b0, b1, b2, b3;
    const unsigned short* pa = lbuf + (waveM * 32 + l16) * LDS_PITCH + lhalf * 8;
    a0.u[0] = *(const uint4*)(pa);
    a0.u[1] = *(const uint4*)(pa + 16);
    a1.u[0] = *(const uint4*)(pa + 16 * LDS_PITCH);
    a1.u[1] = *(const uint4*)(pa + 16 * LDS_PITCH + 16);
    const unsigned short* pb = lbuf + AB_STRIDE + (waveN * 64 + l16) * LDS_PITCH + lhalf * 8;
    b0.u[0] = *(const uint4*)(pb);
    b0.u[1] = *(const uint4*)(pb + 16);
    b1.u[0] = *(const uint4*)(pb + 16 * LDS_PITCH);
    b1.u[1] = *(const uint4*)(pb + 16 * LDS_PITCH + 16);
    b2.u[0] = *(const uint4*)(pb + 32 * LDS_PITCH);
    b2.u[1] = *(const uint4*)(pb + 32 * LDS_PITCH + 16);
    b3.u[0] = *(const uint4*)(pb + 48 * LDS_PITCH);
    b3.u[1] = *(const uint4*)(pb + 48 * LDS_PITCH + 16);
    acc[0][0] = wmma_bf16(a0.v, b0.v, acc[0][0]);
    acc[1][0] = wmma_bf16(a1.v, b0.v, acc[1][0]);
    acc[0][1] = wmma_bf16(a0.v, b1.v, acc[0][1]);
    acc[1][1] = wmma_bf16(a1.v, b1.v, acc[1][1]);
    acc[0][2] = wmma_bf16(a0.v, b2.v, acc[0][2]);
    acc[1][2] = wmma_bf16(a1.v, b2.v, acc[1][2]);
    acc[0][3] = wmma_bf16(a0.v, b3.v, acc[0][3]);
    acc[1][3] = wmma_bf16(a1.v, b3.v, acc[1][3]);
}

// ---------------------------------------------------------------------------
// Kernel 1: x f32 -> bf16 (same layout [8192,1024])
// ---------------------------------------------------------------------------
__global__ __launch_bounds__(256) void convert_x_kernel(const float* __restrict__ src,
                                                        unsigned short* __restrict__ dst,
                                                        int n4) {
    int i = blockIdx.x * blockDim.x + threadIdx.x;
    if (i >= n4) return;
    float4 v = ((const float4*)src)[i];
    unsigned int lo = (unsigned int)f32_to_bf16(v.x) | ((unsigned int)f32_to_bf16(v.y) << 16);
    unsigned int hi = (unsigned int)f32_to_bf16(v.z) | ((unsigned int)f32_to_bf16(v.w) << 16);
    ((uint2*)dst)[i] = make_uint2(lo, hi);
}

// ---------------------------------------------------------------------------
// Kernel 2: transpose + convert: dst[c*rows + r] = bf16(src[r*cols + c])
// ---------------------------------------------------------------------------
__global__ __launch_bounds__(256) void transpose_bf16_kernel(const float* __restrict__ src,
                                                             unsigned short* __restrict__ dst,
                                                             int rows, int cols) {
    __shared__ float tile[32][33];
    const int c0 = blockIdx.x * 32;
    const int r0 = blockIdx.y * 32;
    const int tx = threadIdx.x;
    const int ty = threadIdx.y;
#pragma unroll
    for (int i = 0; i < 32; i += 8)
        tile[ty + i][tx] = src[(size_t)(r0 + ty + i) * cols + (c0 + tx)];
    __syncthreads();
#pragma unroll
    for (int i = 0; i < 32; i += 8)
        dst[(size_t)(c0 + ty + i) * rows + (r0 + tx)] = f32_to_bf16(tile[tx][ty + i]);
}

// ---------------------------------------------------------------------------
// Kernel 3: per-token pattern routing (exact jax.lax.top_k semantics).
// ---------------------------------------------------------------------------
__global__ __launch_bounds__(64) void pattern_gate_kernel(
        const float* __restrict__ pattern_affinity,  // [64, 2048]
        const float* __restrict__ topk_w,            // [8192, 16]
        const int*   __restrict__ topk_idx,          // [8192, 16]
        int*   __restrict__ pidx,                    // [8192, 4]
        float* __restrict__ pw)                      // [8192, 4]
{
    __shared__ float sc[64];
    __shared__ int   sidx[16];
    __shared__ float sw[16];
    const int t = blockIdx.x;
    const int p = threadIdx.x;
    if (p < 16) {
        sidx[p] = topk_idx[t * 16 + p];
        sw[p]   = topk_w[t * 16 + p];
    }
    __syncthreads();
    const float* row = pattern_affinity + p * 2048;
    float s = 0.f;
#pragma unroll
    for (int k = 0; k < 16; ++k) s += row[sidx[k]] * sw[k];
    sc[p] = s;
    __syncthreads();
    if (p == 0) {
        float sc4[4]; int idx4[4];
#pragma unroll
        for (int j = 0; j < 4; ++j) {
            float best = -INFINITY; int bi = 0;
            for (int q = 0; q < 64; ++q)
                if (sc[q] > best) { best = sc[q]; bi = q; }
            sc[bi] = -INFINITY;
            sc4[j] = best; idx4[j] = bi;
        }
        const float mx = sc4[0];
        float e0 = __expf(sc4[0] - mx), e1 = __expf(sc4[1] - mx);
        float e2 = __expf(sc4[2] - mx), e3 = __expf(sc4[3] - mx);
        const float inv = 1.f / (e0 + e1 + e2 + e3);
        pidx[t*4+0] = idx4[0]; pidx[t*4+1] = idx4[1];
        pidx[t*4+2] = idx4[2]; pidx[t*4+3] = idx4[3];
        pw[t*4+0] = e0*inv; pw[t*4+1] = e1*inv; pw[t*4+2] = e2*inv; pw[t*4+3] = e3*inv;
    }
}

// ---------------------------------------------------------------------------
// Kernel 4: up GEMM  h = gelu( (x @ up_w + up_b) * sigmoid(gate_mix) ) -> bf16
// A = xb [8192,1024] bf16, B = upT [4096,1024] bf16 (pre-transposed).
// Block 256 thr = 8 waves (4M x 2N), block tile 128x128, async-LDS staged.
// ---------------------------------------------------------------------------
__global__ __launch_bounds__(256) void up_gemm_kernel(
        const unsigned short* __restrict__ xb,
        const unsigned short* __restrict__ upT,
        const float* __restrict__ up_b,
        const float* __restrict__ gates,    // [64, 4096] f32
        const int*   __restrict__ pidx,     // [8192, 4]
        const float* __restrict__ pw,       // [8192, 4]
        unsigned short* __restrict__ h)     // [8192, 4096] bf16
{
    const int K = 1024, DFF = 4096, NSTEP = K / 32;
    __shared__ unsigned short smem[2 * BUF_STRIDE];

    const int tid   = threadIdx.x;
    const int lane  = tid & 31;
    const int wave  = tid >> 5;
    const int waveM = wave & 3;
    const int waveN = wave >> 2;
    const int l16   = lane & 15;
    const int lhalf = lane >> 4;
    const int blockM = blockIdx.x * 128;
    const int blockN = blockIdx.y * 128;
    const int m0 = blockM + waveM * 32;
    const int n0 = blockN + waveN * 64;

    const unsigned short* gA = xb  + (size_t)blockM * K;
    const unsigned short* gB = upT + (size_t)blockN * K;

    const v8f vzero = {0.f,0.f,0.f,0.f,0.f,0.f,0.f,0.f};
    v8f acc[2][4];
#pragma unroll
    for (int i = 0; i < 2; ++i)
#pragma unroll
        for (int j = 0; j < 4; ++j) acc[i][j] = vzero;

    stage_tile(smem, gA, gB, tid, K, 0);
    int boff = 0;
#pragma clang loop unroll(disable)
    for (int s = 0; s < NSTEP; ++s) {
        wait_async0();          // own async ops for buf done (issued one stage ago)
        __syncthreads();        // tile visible; everyone done reading other buffer
        const int noff = BUF_STRIDE - boff;
        if (s + 1 < NSTEP)
            stage_tile(smem + noff, gA, gB, tid, K, (s + 1) * 32);
        compute_step(acc, smem + boff, waveM, waveN, l16, lhalf);
        boff = noff;
    }

    // Fused epilogue: bias, gate-mix, sigmoid, exact GELU, bf16 store.
    // D layout: lane element r -> row (r + lhalf*8), col l16.
#pragma unroll
    for (int ms = 0; ms < 2; ++ms) {
#pragma unroll
        for (int r = 0; r < 8; ++r) {
            const int t = m0 + ms * 16 + lhalf * 8 + r;
            const int4   pi = *(const int4*)(pidx + t * 4);
            const float4 pv = *(const float4*)(pw + t * 4);
            const float* g0 = gates + (size_t)pi.x * DFF;
            const float* g1 = gates + (size_t)pi.y * DFF;
            const float* g2 = gates + (size_t)pi.z * DFF;
            const float* g3 = gates + (size_t)pi.w * DFF;
            unsigned short* hrow = h + (size_t)t * DFF;
#pragma unroll
            for (int ns = 0; ns < 4; ++ns) {
                const int f = n0 + ns * 16 + l16;
                float H = acc[ms][ns][r] + up_b[f];
                float g = pv.x * g0[f] + pv.y * g1[f] + pv.z * g2[f] + pv.w * g3[f];
                float sgm = 1.f / (1.f + __expf(-g));
                float v = H * sgm;
                float ge = 0.5f * v * (1.f + erff(v * 0.70710678118654752f));
                hrow[f] = f32_to_bf16(ge);
            }
        }
    }
}

// ---------------------------------------------------------------------------
// Kernel 5: down GEMM  out = h @ down_w + down_b  (f32 out)
// A = h [8192,4096] bf16, B = downT [1024,4096] bf16 (pre-transposed).
// ---------------------------------------------------------------------------
__global__ __launch_bounds__(256) void down_gemm_kernel(
        const unsigned short* __restrict__ hb,
        const unsigned short* __restrict__ downT,
        const float* __restrict__ down_b,
        float* __restrict__ out)            // [8192, 1024] f32
{
    const int K = 4096, DM = 1024, NSTEP = K / 32;
    __shared__ unsigned short smem[2 * BUF_STRIDE];

    const int tid   = threadIdx.x;
    const int lane  = tid & 31;
    const int wave  = tid >> 5;
    const int waveM = wave & 3;
    const int waveN = wave >> 2;
    const int l16   = lane & 15;
    const int lhalf = lane >> 4;
    const int blockM = blockIdx.x * 128;
    const int blockN = blockIdx.y * 128;
    const int m0 = blockM + waveM * 32;
    const int n0 = blockN + waveN * 64;

    const unsigned short* gA = hb    + (size_t)blockM * K;
    const unsigned short* gB = downT + (size_t)blockN * K;

    const v8f vzero = {0.f,0.f,0.f,0.f,0.f,0.f,0.f,0.f};
    v8f acc[2][4];
#pragma unroll
    for (int i = 0; i < 2; ++i)
#pragma unroll
        for (int j = 0; j < 4; ++j) acc[i][j] = vzero;

    stage_tile(smem, gA, gB, tid, K, 0);
    int boff = 0;
#pragma clang loop unroll(disable)
    for (int s = 0; s < NSTEP; ++s) {
        wait_async0();
        __syncthreads();
        const int noff = BUF_STRIDE - boff;
        if (s + 1 < NSTEP)
            stage_tile(smem + noff, gA, gB, tid, K, (s + 1) * 32);
        compute_step(acc, smem + boff, waveM, waveN, l16, lhalf);
        boff = noff;
    }

#pragma unroll
    for (int ms = 0; ms < 2; ++ms) {
#pragma unroll
        for (int r = 0; r < 8; ++r) {
            const int row = m0 + ms * 16 + lhalf * 8 + r;
            float* orow = out + (size_t)row * DM;
#pragma unroll
            for (int ns = 0; ns < 4; ++ns) {
                const int n = n0 + ns * 16 + l16;
                orow[n] = acc[ms][ns][r] + down_b[n];
            }
        }
    }
}

// ---------------------------------------------------------------------------
// Host launcher
// ---------------------------------------------------------------------------
extern "C" void kernel_launch(void* const* d_in, const int* in_sizes, int n_in,
                              void* d_out, int out_size, void* d_ws, size_t ws_size,
                              hipStream_t stream) {
    const float* x     = (const float*)d_in[0];
    // d_in[1] = router_out (unused by reference)
    const float* tkw   = (const float*)d_in[2];   // topk_neuron_weights [8192,16]
    const float* paff  = (const float*)d_in[3];   // pattern_affinity    [64,2048]
    const float* gates = (const float*)d_in[4];   // gates               [64,4096]
    const float* up_w  = (const float*)d_in[5];   // [1024,4096]
    const float* up_b  = (const float*)d_in[6];   // [4096]
    const float* dn_w  = (const float*)d_in[7];   // [4096,1024]
    const float* dn_b  = (const float*)d_in[8];   // [1024]
    const int*   tki   = (const int*)d_in[9];     // topk_neuron_idx [8192,16]
    float* out = (float*)d_out;

    const int TOK = 4 * 2048;   // 8192
    const int DM  = 1024;
    const int DFF = 4096;

    char* p = (char*)d_ws;
    auto carve = [&](size_t bytes) -> char* {
        char* r = p;
        p += (bytes + 255) & ~(size_t)255;
        return r;
    };
    unsigned short* xb   = (unsigned short*)carve((size_t)TOK * DM * 2);   // 16 MB
    unsigned short* upT  = (unsigned short*)carve((size_t)DM * DFF * 2);   //  8 MB
    unsigned short* dnT  = (unsigned short*)carve((size_t)DM * DFF * 2);   //  8 MB
    unsigned short* hbuf = (unsigned short*)carve((size_t)TOK * DFF * 2);  // 64 MB
    int*   pidx = (int*)carve((size_t)TOK * 4 * sizeof(int));
    float* pwv  = (float*)carve((size_t)TOK * 4 * sizeof(float));

    convert_x_kernel<<<(TOK * DM / 4 + 255) / 256, 256, 0, stream>>>(x, xb, TOK * DM / 4);
    transpose_bf16_kernel<<<dim3(DFF / 32, DM / 32), dim3(32, 8), 0, stream>>>(up_w, upT, DM, DFF);
    transpose_bf16_kernel<<<dim3(DM / 32, DFF / 32), dim3(32, 8), 0, stream>>>(dn_w, dnT, DFF, DM);
    pattern_gate_kernel<<<TOK, 64, 0, stream>>>(paff, tkw, tki, pidx, pwv);
    up_gemm_kernel<<<dim3(TOK / 128, DFF / 128), 256, 0, stream>>>(xb, upT, up_b, gates, pidx, pwv, hbuf);
    down_gemm_kernel<<<dim3(TOK / 128, DM / 128), 256, 0, stream>>>(hbuf, dnT, dn_b, out);
}
